// Matformer_60833916780662
// MI455X (gfx1250) — matrix operand order repaired and checked by reading, not compiled, not run on hardware
//
#include <hip/hip_runtime.h>
#include <hip/hip_bf16.h>

// ---------------- problem constants ----------------
#define NN   2048      // nodes
#define EE   16384     // edges
#define BB   64        // graphs
#define DD   128       // hidden dim
#define HH   4         // heads
#define CC   128       // per-head channels
#define HC   512       // HH*CC
#define C3   384       // 3*CC
#define LL   5         // layers
#define EPSF 1e-5f
#define INV_SQRT3C 0.05103103630798288f   // 1/sqrt(384)
#define RBF_GAMMA  15.875f                // (128-1)/8.0

typedef __attribute__((ext_vector_type(16))) _Float16 v16h;
typedef __attribute__((ext_vector_type(8)))  float    v8f;

enum { EPI_BIAS = 0, EPI_SOFTPLUS = 1, EPI_GATE = 2, EPI_SILU = 3 };

union Frag { v16h h; uint4 u[2]; };

// ================= WMMA GEMM: C = A(MxK,f16 rowmajor) * Bpacked + bias, epilogue =================
// Block = 128 threads (4 waves). Block tile = 64x64. Wave tile = 16 rows x 64 cols.
// Ping-pong double buffering: prefetch fragments for k+32 while WMMAs for k execute,
// so s_wait_loadcnt only covers the previous stage. K must be a multiple of 64 (always true here).
template <int EPI>
__global__ __launch_bounds__(128) void gemm_wmma(
    const _Float16* __restrict__ A, const _Float16* __restrict__ Bp,
    const float* __restrict__ bias, const _Float16* __restrict__ gate,
    _Float16* __restrict__ C16, float* __restrict__ C32,
    int M, int N, int K)
{
  const int lane = threadIdx.x & 31;
  const int wave = threadIdx.x >> 5;
  const int nBlk = N >> 6;
  const int bx = blockIdx.x % nBlk;
  const int by = blockIdx.x / nBlk;
  const int m0 = by * 64 + wave * 16;
  const int n0 = bx * 64;
  const int KT = K >> 5;

  v8f zero = {0.f,0.f,0.f,0.f,0.f,0.f,0.f,0.f};
  v8f acc[4];
#pragma unroll
  for (int s = 0; s < 4; ++s) acc[s] = zero;

  const int arow = m0 + (lane & 15);
  const int koff = (lane >> 4) * 8;           // A 16x32 f16 lane layout
  const _Float16* abase = A + (size_t)arow * K + koff;

  Frag a0, a1, b0[4], b1[4];
  auto loadFrag = [&](Frag& fa, Frag* fb, int kt) {
    fa.u[0] = *(const uint4*)(abase + kt);
    fa.u[1] = *(const uint4*)(abase + kt + 16);
    const int kti = kt >> 5;
#pragma unroll
    for (int s = 0; s < 4; ++s) {
      const _Float16* bp =
          Bp + ((((size_t)((n0 >> 4) + s) * KT + kti) * 32 + lane) << 4);
      fb[s].u[0] = *(const uint4*)(bp);
      fb[s].u[1] = *(const uint4*)(bp + 8);
    }
  };

  loadFrag(a0, b0, 0);
  for (int kt = 0; kt < K; kt += 64) {
    loadFrag(a1, b1, kt + 32);
#pragma unroll
    for (int s = 0; s < 4; ++s)
      acc[s] = __builtin_amdgcn_wmma_f32_16x16x32_f16(
          false, a0.h, false, b0[s].h, (short)0, acc[s], false, false);
    if (kt + 64 < K) loadFrag(a0, b0, kt + 64);
#pragma unroll
    for (int s = 0; s < 4; ++s)
      acc[s] = __builtin_amdgcn_wmma_f32_16x16x32_f16(
          false, a1.h, false, b1[s].h, (short)0, acc[s], false, false);
  }

  const int rbase = m0 + ((lane >> 4) << 3);  // C/D layout: lanes>=16 hold rows +8
#pragma unroll
  for (int s = 0; s < 4; ++s) {
    const int col = n0 + s * 16 + (lane & 15);
    const float bv = bias ? bias[col] : 0.0f;
#pragma unroll
    for (int g = 0; g < 8; ++g) {
      const int r = rbase + g;
      float v = acc[s][g] + bv;
      if (EPI == EPI_SOFTPLUS) v = (v > 20.f) ? v : log1pf(__expf(v));
      if (EPI == EPI_SILU)     v = v / (1.f + __expf(-v));
      const size_t idx = (size_t)r * N + col;
      if (EPI == EPI_GATE)     v *= (float)gate[idx];
      if (C16) C16[idx] = (_Float16)v;
      if (C32) C32[idx] = v;
    }
  }
}

// Pack f32 weight (KxN rowmajor) into WMMA B-fragment f16 layout.
__global__ void pack_b(const float* __restrict__ B, _Float16* __restrict__ Bp, int K, int N) {
  int i = blockIdx.x * blockDim.x + threadIdx.x;
  if (i >= K * N) return;
  int k = i / N, n = i - k * N;
  int nti = n >> 4, kti = k >> 5;
  int lane = (n & 15) | (((k >> 3) & 1) << 4);
  int half = (k & 7) | (((k >> 4) & 1) << 3);
  size_t idx = ((((size_t)nti * (K >> 5) + kti) * 32 + lane) << 4) + half;
  Bp[idx] = (_Float16)B[(size_t)k * N + n];
}

// ============ fused edge-message kernel =============
// One wave (32 threads) owns 16 edge-head rows.
// Phase 0: gather q/k/v/e, alpha-LN -> sigmoid gate, build m_in  (LDS, f16)
// Phase 1: t1 = (m_in @ wm + bm) * gate   (WMMA, A from LDS, B packed from L2) -> LDS
// Phase 2: m  = t1 @ wmsg + bmsg          (WMMA, A from LDS)
// Phase 3: row-LN of m via half-wave shfl reductions, atomicAdd segment-sum into agg.
__global__ __launch_bounds__(32) void edge_fused(
    const int* __restrict__ ei, const float* __restrict__ q32,
    const float* __restrict__ k32, const float* __restrict__ v32,
    const float* __restrict__ e32,
    const float* __restrict__ lag, const float* __restrict__ lab,
    const _Float16* __restrict__ wmP,   const float* __restrict__ bm,
    const _Float16* __restrict__ wmsgP, const float* __restrict__ bmsg,
    const float* __restrict__ lmg, const float* __restrict__ lmb,
    float* __restrict__ agg)
{
  __shared__ _Float16 sMin[16 * C3];
  __shared__ _Float16 sGate[16 * C3];
  __shared__ _Float16 sT1[16 * C3];

  const int lane = threadIdx.x;
  const int row0 = blockIdx.x * 16;     // global edge-head row base

  // ---------- phase 0: build m_in + gate for 16 rows ----------
  for (int r = 0; r < 16; ++r) {
    const int rowId = row0 + r;
    const int e = rowId >> 2, h = rowId & 3;
    const int s = ei[e], d = ei[EE + e];
    const float* qrow = q32 + ((size_t)d * HH + h) * CC;
    const float* kdst = k32 + ((size_t)d * HH + h) * CC;
    const float* ksrc = k32 + ((size_t)s * HH + h) * CC;
    const float* vdst = v32 + ((size_t)d * HH + h) * CC;
    const float* vsrc = v32 + ((size_t)s * HH + h) * CC;
    const float* erow = e32 + ((size_t)e * HH + h) * CC;

    float a[12], mv[12];
    float sum = 0.f, sq = 0.f;
#pragma unroll
    for (int t = 0; t < 12; ++t) {
      int j = lane + t * 32;
      int c = j & 127;
      float qv = qrow[c], kv, m;
      if (j < 128)      { kv = kdst[c]; m = vdst[c]; }
      else if (j < 256) { kv = ksrc[c]; m = vsrc[c]; }
      else              { kv = erow[c]; m = erow[c]; }
      float al = qv * kv * INV_SQRT3C;
      a[t] = al; mv[t] = m;
      sum += al; sq += al * al;
    }
    for (int off = 16; off; off >>= 1) {
      sum += __shfl_xor(sum, off, 32);
      sq  += __shfl_xor(sq,  off, 32);
    }
    float mu = sum * (1.f / (float)C3);
    float var = sq * (1.f / (float)C3) - mu * mu;
    float rs = rsqrtf(var + EPSF);
#pragma unroll
    for (int t = 0; t < 12; ++t) {
      int j = lane + t * 32;
      float ln = (a[t] - mu) * rs * lag[j] + lab[j];
      float g = 1.f / (1.f + __expf(-ln));
      sGate[r * C3 + j] = (_Float16)g;
      sMin[r * C3 + j]  = (_Float16)mv[t];
    }
  }
  __syncthreads();

  const int koff = (lane >> 4) * 8;
  const int rb = ((lane >> 4) << 3);
  const int arowLoc = (lane & 15);

  // ---------- phase 1: t1 = (m_in @ wm + bm) * gate ----------
  for (int grp = 0; grp < 6; ++grp) {       // 6 groups of 64 cols (N=384)
    v8f zero = {0.f,0.f,0.f,0.f,0.f,0.f,0.f,0.f};
    v8f acc[4];
#pragma unroll
    for (int s = 0; s < 4; ++s) acc[s] = zero;

    for (int kt = 0; kt < C3; kt += 32) {
      Frag ua;
      ua.u[0] = *(const uint4*)(&sMin[arowLoc * C3 + koff + kt]);
      ua.u[1] = *(const uint4*)(&sMin[arowLoc * C3 + koff + kt + 16]);
      const int kti = kt >> 5;
#pragma unroll
      for (int s = 0; s < 4; ++s) {
        const int nti = grp * 4 + s;
        const _Float16* bp = wmP + ((((size_t)nti * 12 + kti) * 32 + lane) << 4);
        Frag ub;
        ub.u[0] = *(const uint4*)(bp);
        ub.u[1] = *(const uint4*)(bp + 8);
        acc[s] = __builtin_amdgcn_wmma_f32_16x16x32_f16(
            false, ua.h, false, ub.h, (short)0, acc[s], false, false);
      }
    }
#pragma unroll
    for (int s = 0; s < 4; ++s) {
      const int col = grp * 64 + s * 16 + (lane & 15);
      const float bb = bm[col];
#pragma unroll
      for (int g = 0; g < 8; ++g) {
        const int rl = rb + g;
        float v = acc[s][g] + bb;
        v *= (float)sGate[rl * C3 + col];
        sT1[rl * C3 + col] = (_Float16)v;
      }
    }
  }
  __syncthreads();

  // ---------- phase 2: m = t1 @ wmsg + bmsg (keep full 128-wide row in regs) ----------
  v8f zero = {0.f,0.f,0.f,0.f,0.f,0.f,0.f,0.f};
  v8f acc2[8];
#pragma unroll
  for (int s = 0; s < 8; ++s) acc2[s] = zero;

  for (int kt = 0; kt < C3; kt += 32) {
    Frag ua;
    ua.u[0] = *(const uint4*)(&sT1[arowLoc * C3 + koff + kt]);
    ua.u[1] = *(const uint4*)(&sT1[arowLoc * C3 + koff + kt + 16]);
    const int kti = kt >> 5;
#pragma unroll
    for (int s = 0; s < 8; ++s) {
      const _Float16* bp = wmsgP + ((((size_t)s * 12 + kti) * 32 + lane) << 4);
      Frag ub;
      ub.u[0] = *(const uint4*)(bp);
      ub.u[1] = *(const uint4*)(bp + 8);
      acc2[s] = __builtin_amdgcn_wmma_f32_16x16x32_f16(
          false, ua.h, false, ub.h, (short)0, acc2[s], false, false);
    }
  }

  // ---------- phase 3: row LayerNorm + atomic segment sum ----------
  // Row (rb+g) lives in one 16-lane half; xor offsets 1,2,4,8 reduce exactly that row.
#pragma unroll
  for (int g = 0; g < 8; ++g) {
    float sum = 0.f, sq = 0.f;
#pragma unroll
    for (int s = 0; s < 8; ++s) {
      float v = acc2[s][g] + bmsg[s * 16 + (lane & 15)];
      sum += v; sq += v * v;
    }
    for (int off = 8; off; off >>= 1) {
      sum += __shfl_xor(sum, off, 32);
      sq  += __shfl_xor(sq,  off, 32);
    }
    float mu = sum * (1.f / (float)CC);
    float var = sq * (1.f / (float)CC) - mu * mu;
    float rs = rsqrtf(var + EPSF);

    const int rowId = row0 + rb + g;
    const int e = rowId >> 2, h = rowId & 3;
    const int d = ei[EE + e];
    float* ag = agg + ((size_t)d * HH + h) * CC;
#pragma unroll
    for (int s = 0; s < 8; ++s) {
      const int col = s * 16 + (lane & 15);
      float v = acc2[s][g] + bmsg[col];
      float y = (v - mu) * rs * lmg[col] + lmb[col];
      atomicAdd(&ag[col], y);
    }
  }
}

// ================= elementwise / small kernels =================
__global__ void embed_kernel(const float* __restrict__ x, const float* __restrict__ w,
                             const float* __restrict__ b, float* __restrict__ h32,
                             _Float16* __restrict__ h16) {
  int i = blockIdx.x * blockDim.x + threadIdx.x;
  if (i >= NN * DD) return;
  int n = i >> 7, c = i & 127;
  float s = b[c];
  const float* xr = x + (size_t)n * 92;
  for (int f = 0; f < 92; ++f) s += xr[f] * w[f * DD + c];
  h32[i] = s; h16[i] = (_Float16)s;
}

__global__ void rbf_kernel(const float* __restrict__ ea, _Float16* __restrict__ rbf16) {
  int i = blockIdx.x * blockDim.x + threadIdx.x;
  if (i >= EE * 128) return;
  int e = i >> 7, bin = i & 127;
  float a0 = ea[e * 3], a1 = ea[e * 3 + 1], a2 = ea[e * 3 + 2];
  float d = sqrtf(a0 * a0 + a1 * a1 + a2 * a2);
  float c = 8.f * (float)bin / 127.f;
  float t = d - c;
  rbf16[i] = (_Float16)__expf(-RBF_GAMMA * t * t);
}

__global__ void f32_to_f16(const float* __restrict__ in, _Float16* __restrict__ out, int n) {
  int i = blockIdx.x * blockDim.x + threadIdx.x;
  if (i < n) out[i] = (_Float16)in[i];
}

__global__ void bn_stats(const float* __restrict__ out32, float* __restrict__ mu,
                         float* __restrict__ var) {
  __shared__ float ss[256], qq[256];
  int c = blockIdx.x, t = threadIdx.x;
  float s = 0.f, q = 0.f;
  for (int r = t; r < NN; r += 256) {
    float v = out32[(size_t)r * DD + c];
    s += v; q += v * v;
  }
  ss[t] = s; qq[t] = q; __syncthreads();
  for (int o = 128; o; o >>= 1) {
    if (t < o) { ss[t] += ss[t + o]; qq[t] += qq[t + o]; }
    __syncthreads();
  }
  if (t == 0) {
    float m = ss[0] / (float)NN;
    mu[c] = m; var[c] = qq[0] / (float)NN - m * m;
  }
}

__global__ void bn_silu_skip(const float* __restrict__ out32, const float* __restrict__ mu,
                             const float* __restrict__ var, const float* __restrict__ g,
                             const float* __restrict__ b, const float* __restrict__ skip32,
                             float* __restrict__ h32, _Float16* __restrict__ h16) {
  int i = blockIdx.x * blockDim.x + threadIdx.x;
  if (i >= NN * DD) return;
  int c = i & 127;
  float v = (out32[i] - mu[c]) * rsqrtf(var[c] + EPSF) * g[c] + b[c];
  v = v / (1.f + __expf(-v));
  float hn = v + skip32[i];
  h32[i] = hn; h16[i] = (_Float16)hn;
}

__global__ void head_kernel(const float* __restrict__ feat32, const float* __restrict__ fcow,
                            const float* __restrict__ fcob, const int* __restrict__ batch,
                            float* __restrict__ tmp, float* __restrict__ qsum,
                            float* __restrict__ cnt) {
  int n = blockIdx.x * blockDim.x + threadIdx.x;
  if (n >= NN) return;
  const float* fr = feat32 + (size_t)n * DD;
  float ea = fcob[0], qa = fcob[1];
  for (int c = 0; c < DD; ++c) { ea += fr[c] * fcow[c * 2]; qa += fr[c] * fcow[c * 2 + 1]; }
  tmp[n * 2] = ea; tmp[n * 2 + 1] = qa;
  atomicAdd(&qsum[batch[n]], qa);
  atomicAdd(&cnt[batch[n]], 1.0f);
}

__global__ void fix_kernel(const float* __restrict__ tmp, const int* __restrict__ batch,
                           const float* __restrict__ qsum, const float* __restrict__ cnt,
                           float* __restrict__ out) {
  int n = blockIdx.x * blockDim.x + threadIdx.x;
  if (n >= NN) return;
  int b = batch[n];
  out[n * 2] = tmp[n * 2];
  out[n * 2 + 1] = tmp[n * 2 + 1] - qsum[b] / cnt[b];
}

// ================= host orchestration =================
extern "C" void kernel_launch(void* const* d_in, const int* in_sizes, int n_in,
                              void* d_out, int out_size, void* d_ws, size_t ws_size,
                              hipStream_t stream) {
  (void)in_sizes; (void)n_in; (void)out_size; (void)ws_size;

  const float* x      = (const float*)d_in[0];
  const float* eattr  = (const float*)d_in[1];
  const int*   ei     = (const int*)d_in[2];
  const int*   batch  = (const int*)d_in[3];
  const float* emb_w  = (const float*)d_in[4];
  const float* emb_b  = (const float*)d_in[5];
  const float* rbf_w1 = (const float*)d_in[6];
  const float* rbf_b1 = (const float*)d_in[7];
  const float* rbf_w2 = (const float*)d_in[8];
  const float* rbf_b2 = (const float*)d_in[9];
  const float* wq = (const float*)d_in[10]; const float* bq = (const float*)d_in[11];
  const float* wk = (const float*)d_in[12]; const float* bk = (const float*)d_in[13];
  const float* wv = (const float*)d_in[14]; const float* bv = (const float*)d_in[15];
  const float* we = (const float*)d_in[16];
  const float* wcat = (const float*)d_in[17]; const float* bcat = (const float*)d_in[18];
  const float* wm = (const float*)d_in[19];   const float* bm = (const float*)d_in[20];
  const float* lag = (const float*)d_in[21];  const float* lab = (const float*)d_in[22];
  const float* wmsg = (const float*)d_in[23]; const float* bmsg = (const float*)d_in[24];
  const float* lmg = (const float*)d_in[25];  const float* lmb = (const float*)d_in[26];
  const float* bng = (const float*)d_in[27];  const float* bnb = (const float*)d_in[28];
  const float* wskip = (const float*)d_in[29]; const float* bskip = (const float*)d_in[30];
  const float* fc_w = (const float*)d_in[31];  const float* fc_b = (const float*)d_in[32];
  const float* fco_w = (const float*)d_in[33]; const float* fco_b = (const float*)d_in[34];
  float* out = (float*)d_out;

  // ---- workspace carve-up ----
  char* p = (char*)d_ws;
  auto alloc = [&](size_t bytes) -> void* {
    void* r = (void*)p;
    p += (bytes + 255) & ~(size_t)255;
    return r;
  };
  const size_t ER = (size_t)EE * HH;  // 65536 edge-head rows

  float*    h32   = (float*)   alloc((size_t)NN * DD * 4);
  _Float16* h16   = (_Float16*)alloc((size_t)NN * DD * 2);
  _Float16* rbf16 = (_Float16*)alloc((size_t)EE * 128 * 2);
  _Float16* r116  = (_Float16*)alloc((size_t)EE * 128 * 2);
  _Float16* ef16  = (_Float16*)alloc((size_t)EE * 128 * 2);
  float*    q32   = (float*)   alloc((size_t)NN * HC * 4);
  float*    k32   = (float*)   alloc((size_t)NN * HC * 4);
  float*    v32   = (float*)   alloc((size_t)NN * HC * 4);
  float*    e32   = (float*)   alloc((size_t)EE * HC * 4);
  float*    agg32 = (float*)   alloc((size_t)NN * HC * 4);
  _Float16* agg16 = (_Float16*)alloc((size_t)NN * HC * 2);
  float*    out32 = (float*)   alloc((size_t)NN * DD * 4);
  float*    skip32= (float*)   alloc((size_t)NN * DD * 4);
  float*    mu    = (float*)   alloc(DD * 4);
  float*    var   = (float*)   alloc(DD * 4);
  float*    feat32= (float*)   alloc((size_t)NN * DD * 4);
  float*    htmp  = (float*)   alloc((size_t)NN * 2 * 4);
  float*    qsum  = (float*)   alloc(BB * 4);
  float*    cnt   = (float*)   alloc(BB * 4);

  _Float16* rbfw1p = (_Float16*)alloc((size_t)128 * 128 * 2);
  _Float16* rbfw2p = (_Float16*)alloc((size_t)128 * 128 * 2);
  _Float16* fcwp   = (_Float16*)alloc((size_t)128 * 128 * 2);
  _Float16 *wqp[LL], *wkp[LL], *wvp[LL], *wep[LL], *wmp[LL], *wmsgp[LL], *wcatp[LL], *wskipp[LL];
  for (int l = 0; l < LL; ++l) {
    wqp[l]    = (_Float16*)alloc((size_t)DD * HC * 2);
    wkp[l]    = (_Float16*)alloc((size_t)DD * HC * 2);
    wvp[l]    = (_Float16*)alloc((size_t)DD * HC * 2);
    wep[l]    = (_Float16*)alloc((size_t)DD * HC * 2);
    wmp[l]    = (_Float16*)alloc((size_t)C3 * C3 * 2);
    wmsgp[l]  = (_Float16*)alloc((size_t)C3 * CC * 2);
    wcatp[l]  = (_Float16*)alloc((size_t)HC * CC * 2);
    wskipp[l] = (_Float16*)alloc((size_t)DD * CC * 2);
  }

  auto packW = [&](const float* W, _Float16* Wp, int K, int N) {
    int tot = K * N;
    pack_b<<<(tot + 255) / 256, 256, 0, stream>>>(W, Wp, K, N);
  };
  auto gemm = [&](int epi, const _Float16* A, const _Float16* Bp, const float* bias,
                  const _Float16* gate, _Float16* C16, float* C32, int M, int N, int K) {
    dim3 grid((unsigned)((M / 64) * (N / 64)));
    switch (epi) {
      case EPI_BIAS:     gemm_wmma<EPI_BIAS><<<grid, 128, 0, stream>>>(A, Bp, bias, gate, C16, C32, M, N, K); break;
      case EPI_SOFTPLUS: gemm_wmma<EPI_SOFTPLUS><<<grid, 128, 0, stream>>>(A, Bp, bias, gate, C16, C32, M, N, K); break;
      case EPI_GATE:     gemm_wmma<EPI_GATE><<<grid, 128, 0, stream>>>(A, Bp, bias, gate, C16, C32, M, N, K); break;
      case EPI_SILU:     gemm_wmma<EPI_SILU><<<grid, 128, 0, stream>>>(A, Bp, bias, gate, C16, C32, M, N, K); break;
    }
  };

  // ---- pack all weights to WMMA fragment layout ----
  packW(rbf_w1, rbfw1p, 128, 128);
  packW(rbf_w2, rbfw2p, 128, 128);
  packW(fc_w,   fcwp,   128, 128);
  for (int l = 0; l < LL; ++l) {
    packW(wq + (size_t)l * DD * HC, wqp[l], DD, HC);
    packW(wk + (size_t)l * DD * HC, wkp[l], DD, HC);
    packW(wv + (size_t)l * DD * HC, wvp[l], DD, HC);
    packW(we + (size_t)l * DD * HC, wep[l], DD, HC);
    packW(wm + (size_t)l * C3 * C3, wmp[l], C3, C3);
    packW(wmsg + (size_t)l * C3 * CC, wmsgp[l], C3, CC);
    packW(wcat + (size_t)l * HC * CC, wcatp[l], HC, CC);
    packW(wskip + (size_t)l * DD * CC, wskipp[l], DD, CC);
  }

  // ---- embedding + RBF MLP ----
  embed_kernel<<<(NN * DD + 255) / 256, 256, 0, stream>>>(x, emb_w, emb_b, h32, h16);
  rbf_kernel<<<(EE * 128 + 255) / 256, 256, 0, stream>>>(eattr, rbf16);
  gemm(EPI_SOFTPLUS, rbf16, rbfw1p, rbf_b1, nullptr, r116, nullptr, EE, 128, 128);
  gemm(EPI_BIAS,     r116,  rbfw2p, rbf_b2, nullptr, ef16, nullptr, EE, 128, 128);

  // ---- layers ----
  for (int l = 0; l < LL; ++l) {
    gemm(EPI_BIAS, h16,  wqp[l], bq + l * HC, nullptr, nullptr, q32, NN, HC, DD);
    gemm(EPI_BIAS, h16,  wkp[l], bk + l * HC, nullptr, nullptr, k32, NN, HC, DD);
    gemm(EPI_BIAS, h16,  wvp[l], bv + l * HC, nullptr, nullptr, v32, NN, HC, DD);
    gemm(EPI_BIAS, ef16, wep[l], nullptr,     nullptr, nullptr, e32, EE, HC, DD);

    hipMemsetAsync(agg32, 0, (size_t)NN * HC * 4, stream);
    edge_fused<<<(int)(ER / 16), 32, 0, stream>>>(
        ei, q32, k32, v32, e32,
        lag + l * C3, lab + l * C3,
        wmp[l], bm + l * C3,
        wmsgp[l], bmsg + l * CC,
        lmg + l * CC, lmb + l * CC, agg32);

    f32_to_f16<<<(NN * HC + 255) / 256, 256, 0, stream>>>(agg32, agg16, NN * HC);
    gemm(EPI_BIAS, agg16, wcatp[l],  bcat + l * CC,  nullptr, nullptr, out32,  NN, CC, HC);
    gemm(EPI_BIAS, h16,   wskipp[l], bskip + l * CC, nullptr, nullptr, skip32, NN, CC, DD);

    bn_stats<<<DD, 256, 0, stream>>>(out32, mu, var);
    bn_silu_skip<<<(NN * DD + 255) / 256, 256, 0, stream>>>(out32, mu, var,
        bng + l * CC, bnb + l * CC, skip32, h32, h16);
  }

  // ---- output head ----
  gemm(EPI_SILU, h16, fcwp, fc_b, nullptr, nullptr, feat32, NN, DD, DD);
  hipMemsetAsync(qsum, 0, BB * 4, stream);
  hipMemsetAsync(cnt,  0, BB * 4, stream);
  head_kernel<<<(NN + 255) / 256, 256, 0, stream>>>(feat32, fco_w, fco_b, batch, htmp, qsum, cnt);
  fix_kernel<<<(NN + 255) / 256, 256, 0, stream>>>(htmp, batch, qsum, cnt, out);
}
